// LinearAttentionReplacement_14886356648283
// MI455X (gfx1250) — matrix operand order repaired and checked by reading, not compiled, not run on hardware
//
#include <hip/hip_runtime.h>

// Linear attention: out = causal_mask(Q K^T * scale + bias) @ V
// B=2, H=16, S=2048, D=128, all fp32. bf16 WMMA with fp32 accumulation.
// Bias (the dominant HBM stream) is prefetched into registers in WMMA C-layout
// at the top of each k-iteration so its latency hides under K/V staging + GEMM1.

#define S_LEN 2048
#define DHEAD 128
#define BM 64          // query rows per block
#define BN 64          // key rows per k-iteration
#define LDQ 136        // lds stride for Q/K tiles (128 + 8 pad, bf16 elems)
#define LDV 72         // lds stride for V^T and P tiles (64 + 8 pad)

typedef __attribute__((ext_vector_type(16))) __bf16 v16bf;
typedef __attribute__((ext_vector_type(4)))  __bf16 v4bf;
typedef __attribute__((ext_vector_type(8)))  float  v8f;

__device__ __forceinline__ __bf16 f2bf(float f) {
    unsigned int u = __float_as_uint(f);
    u += 0x7FFFu + ((u >> 16) & 1u);          // round-to-nearest-even
    unsigned short h = (unsigned short)(u >> 16);
    return __builtin_bit_cast(__bf16, h);
}

// A-fragment (16x32 bf16, ISA 7.12.2 layout): lds is row-major [row][k].
// lane L: m = L&15, half = L>>4. VGPR v holds K = (v<4?0:16) + half*8 + (v%4)*2, +1.
__device__ __forceinline__ v16bf ldsA(const __bf16* lds, int stride, int row0, int k0, int lane) {
    const int m = lane & 15, half = lane >> 4;
    const __bf16* p = lds + (row0 + m) * stride + k0;
    v16bf a;
#pragma unroll
    for (int v = 0; v < 8; ++v) {
        const int k = ((v < 4) ? 0 : 16) + half * 8 + (v & 3) * 2;
        a[2 * v]     = p[k];
        a[2 * v + 1] = p[k + 1];
    }
    return a;
}

// B-fragment (32x16 bf16): B[k][n] where lds stores row n at [nrow0+n][k] (B = lds^T).
// lane L: n = L&15; elements i = K = k0 + (L>>4)*8 + (i<8 ? i : 16 + i - 8)  (matches ldsA).
__device__ __forceinline__ v16bf ldsBt(const __bf16* lds, int stride, int nrow0, int k0, int lane) {
    const int n = lane & 15;
    const __bf16* p = lds + (nrow0 + n) * stride + k0 + (lane >> 4) * 8;
    v16bf b;
#pragma unroll
    for (int i = 0; i < 16; ++i) b[i] = p[(i < 8) ? i : (i + 8)];
    return b;
}

#define WMMA_BF16(A, Bm, C) \
    __builtin_amdgcn_wmma_f32_16x16x32_bf16(false, (A), false, (Bm), (short)0, (C), false, false)

__global__ __launch_bounds__(256) void linattn_wmma_kernel(
    const float* __restrict__ Q, const float* __restrict__ K, const float* __restrict__ V,
    const float* __restrict__ Bias, const float* __restrict__ Scale,
    const int* __restrict__ CausalP, const int* __restrict__ HasBiasP,
    float* __restrict__ Out)
{
    __shared__ __bf16 sQ[BM * LDQ];      // [q_row][d]
    __shared__ __bf16 sK[BN * LDQ];      // [k_row][d]
    __shared__ __bf16 sVt[DHEAD * LDV];  // [d][k_row]  (transposed on staging)
    __shared__ __bf16 sP[BM * LDV];      // [q_row][k_row] bf16 scores

    const int tid  = threadIdx.x;
    const int lane = tid & 31;
    const int wave = tid >> 5;
    const int qb = blockIdx.x;
    const int bh = blockIdx.y;
    const int qbase = qb * BM;

    const float scale   = Scale[0];
    const int  causal   = CausalP[0];
    const int  has_bias = HasBiasP[0];

    const size_t base_qkv  = (size_t)bh * S_LEN * DHEAD;
    const size_t base_bias = (size_t)bh * S_LEN * S_LEN;

    // ---- stage Q tile once (vectorized f32x4 loads, packed bf16x4 LDS stores) ----
#pragma unroll
    for (int i = 0; i < 8; ++i) {
        const int idx = tid + i * 256;            // float4 index within 64x128 tile
        const int r = idx >> 5;                   // row
        const int c = (idx & 31) * 4;             // col
        const float4 qv = *(const float4*)&Q[base_qkv + (size_t)(qbase + r) * DHEAD + c];
        *(v4bf*)&sQ[r * LDQ + c] = v4bf{f2bf(qv.x), f2bf(qv.y), f2bf(qv.z), f2bf(qv.w)};
    }

    const int mi    = wave >> 1;   // 0..3 : M-subtile of this wave
    const int nhalf = wave & 1;    // 0/1  : which half of N/D columns
    const int ncol  = lane & 15;
    const int half  = lane >> 4;

    v8f o[4] = {v8f{}, v8f{}, v8f{}, v8f{}};   // O accumulators: cols nhalf*64 + j*16

    const int kb_end = causal ? qb : ((int)gridDim.x - 1);
    for (int kb = 0; kb <= kb_end; ++kb) {
        const int kbase = kb * BN;

        // ---- prefetch this tile's bias into registers, WMMA C-layout ----
        // Issued before any LDS dependency: overlaps K/V staging and GEMM1.
        float breg[16];
#pragma unroll
        for (int t = 0; t < 2; ++t) {
            const int kg = kbase + (nhalf * 2 + t) * 16 + ncol;
#pragma unroll
            for (int r = 0; r < 8; ++r) {
                const int mg = qbase + mi * 16 + r + half * 8;
                breg[t * 8 + r] = Bias[base_bias + (size_t)mg * S_LEN + kg];
            }
        }

        __syncthreads();   // previous iteration's LDS reads complete (covers Q staging too)

        // ---- stage K tile and V^T tile ----
#pragma unroll
        for (int i = 0; i < 8; ++i) {
            const int idx = tid + i * 256;
            const int r = idx >> 5;
            const int c = (idx & 31) * 4;
            const size_t g = base_qkv + (size_t)(kbase + r) * DHEAD + c;
            const float4 kv = *(const float4*)&K[g];
            *(v4bf*)&sK[r * LDQ + c] = v4bf{f2bf(kv.x), f2bf(kv.y), f2bf(kv.z), f2bf(kv.w)};
            const float4 vv = *(const float4*)&V[g];
            sVt[(c + 0) * LDV + r] = f2bf(vv.x);
            sVt[(c + 1) * LDV + r] = f2bf(vv.y);
            sVt[(c + 2) * LDV + r] = f2bf(vv.z);
            sVt[(c + 3) * LDV + r] = f2bf(vv.w);
        }
        __syncthreads();

        // ---- GEMM 1: S[64x64] = Q_tile (64x128) x K_tile^T, fp32 acc ----
        v8f s0 = {}, s1 = {};
#pragma unroll
        for (int kk = 0; kk < 4; ++kk) {
            const v16bf a  = ldsA (sQ, LDQ, mi * 16, kk * 32, lane);
            const v16bf b0 = ldsBt(sK, LDQ, (nhalf * 2 + 0) * 16, kk * 32, lane);
            const v16bf b1 = ldsBt(sK, LDQ, (nhalf * 2 + 1) * 16, kk * 32, lane);
            s0 = WMMA_BF16(a, b0, s0);
            s1 = WMMA_BF16(a, b1, s1);
        }

        // ---- epilogue: scale, +bias (from regs), causal zero-mask; P -> LDS bf16 ----
#pragma unroll
        for (int t = 0; t < 2; ++t) {
            v8f& s = t ? s1 : s0;
            const int nloc = (nhalf * 2 + t) * 16 + ncol;
            const int kg = kbase + nloc;
#pragma unroll
            for (int r = 0; r < 8; ++r) {
                const int ml = mi * 16 + r + half * 8;   // C-layout: VGPR r, lane-half selects +8
                const int mg = qbase + ml;
                float sv = s[r] * scale;
                if (has_bias) sv += breg[t * 8 + r];
                if (causal && (kg > mg)) sv = 0.0f;
                sP[ml * LDV + nloc] = f2bf(sv);
            }
        }
        __syncthreads();

        // ---- GEMM 2: O[64x128] += P (64x64) x V_tile (64x128) ----
#pragma unroll
        for (int kk = 0; kk < 2; ++kk) {
            const v16bf a = ldsA(sP, LDV, mi * 16, kk * 32, lane);
#pragma unroll
            for (int j = 0; j < 4; ++j) {
                const v16bf b = ldsBt(sVt, LDV, nhalf * 64 + j * 16, kk * 32, lane);
                o[j] = WMMA_BF16(a, b, o[j]);
            }
        }
    }

    // ---- write O (fp32) ----
#pragma unroll
    for (int j = 0; j < 4; ++j) {
        const int dg = nhalf * 64 + j * 16 + ncol;
#pragma unroll
        for (int r = 0; r < 8; ++r) {
            const int mg = qbase + mi * 16 + r + half * 8;
            Out[((size_t)bh * S_LEN + mg) * DHEAD + dg] = o[j][r];
        }
    }
}

extern "C" void kernel_launch(void* const* d_in, const int* in_sizes, int n_in,
                              void* d_out, int out_size, void* d_ws, size_t ws_size,
                              hipStream_t stream) {
    const float* Q    = (const float*)d_in[0];
    const float* K    = (const float*)d_in[1];
    const float* V    = (const float*)d_in[2];
    const float* Bias = (const float*)d_in[3];
    const float* Sc   = (const float*)d_in[4];
    const int* causal   = (const int*)d_in[5];
    const int* has_bias = (const int*)d_in[6];
    float* Out = (float*)d_out;

    const int BH = 2 * 16;
    dim3 grid(S_LEN / BM, BH);
    dim3 block(256);
    linattn_wmma_kernel<<<grid, block, 0, stream>>>(Q, K, V, Bias, Sc, causal, has_bias, Out);
}
